// ODERGRU_1159641169944
// MI455X (gfx1250) — compile-verified, matrix operand order checked by reading, not run on hardware
//
#include <hip/hip_runtime.h>
#include <hip/hip_bf16.h>

// ---------------------------------------------------------------------------
// CDNA5 (gfx1250) ODE-RGRU pipeline.
// All matmuls (convs / Gram / scan) use v_wmma_f32_16x16x32_f16 (wave32 WMMA).
// f16 operands, f32 accumulation. Conv inputs are spatially pre-padded
// (3-pixel zero halo); the tap loop packs 32/CP kw-adjacent taps per WMMA so
// the full K=32 of every WMMA is utilized, with compile-time load offsets.
// ---------------------------------------------------------------------------

typedef __attribute__((ext_vector_type(16))) _Float16 v16h;
typedef __attribute__((ext_vector_type(8)))  _Float16 v8h;
typedef __attribute__((ext_vector_type(8)))  float    v8f;

__device__ __forceinline__ v8f wmma_f16(v16h a, v16h b, v8f c) {
  // D = A(16x32) * B(32x16) + C(16x16), f32 accum
  return __builtin_amdgcn_wmma_f32_16x16x32_f16(false, a, false, b, (short)0, c,
                                                false, false);
}

__device__ __forceinline__ float sigmoidf_(float x) { return 1.0f / (1.0f + expf(-x)); }
__device__ __forceinline__ float softplusf_(float x) {
  return (x > 0.0f) ? (x + log1pf(expf(-x))) : log1pf(expf(x));
}

// ---------------------------------------------------------------------------
// Fragment builders (16-bit A layout: lanes 0-15 hold M=0-15, K in {0..7,16..23};
// lanes 16-31 hold M=0-15, K in {8..15,24..31}. B symmetric with N in place
// of M). f32 sources converted to f16 on the fly, zero-padded beyond Kv/Nv.
// ---------------------------------------------------------------------------
__device__ __forceinline__ v16h fragA_f32(const float* A, int lda, int kb, int Kv) {
  int lane  = threadIdx.x & 31;
  int idx   = lane & 15;                         // row m
  int base0 = kb + ((lane & 16) ? 8 : 0);
  int base1 = kb + 16 + ((lane & 16) ? 8 : 0);
  union { v16h v; _Float16 e[16]; } a;
#pragma unroll
  for (int j = 0; j < 8; ++j) {
    int k0 = base0 + j, k1 = base1 + j;
    a.e[j]     = (k0 < Kv) ? (_Float16)A[idx * lda + k0] : (_Float16)0.0f;
    a.e[8 + j] = (k1 < Kv) ? (_Float16)A[idx * lda + k1] : (_Float16)0.0f;
  }
  return a.v;
}

__device__ __forceinline__ v16h fragB_f32(const float* B, int ldb, int n0, int kb,
                                          int Kv, int Nv, bool absB) {
  int lane  = threadIdx.x & 31;
  int n     = n0 + (lane & 15);                  // column n
  int base0 = kb + ((lane & 16) ? 8 : 0);
  int base1 = kb + 16 + ((lane & 16) ? 8 : 0);
  union { v16h v; _Float16 e[16]; } b;
#pragma unroll
  for (int j = 0; j < 8; ++j) {
    int k0 = base0 + j, k1 = base1 + j;
    float v0 = (k0 < Kv && n < Nv) ? B[(size_t)k0 * ldb + n] : 0.0f;
    float v1 = (k1 < Kv && n < Nv) ? B[(size_t)k1 * ldb + n] : 0.0f;
    if (absB) { v0 = fabsf(v0); v1 = fabsf(v1); }
    b.e[j] = (_Float16)v0;
    b.e[8 + j] = (_Float16)v1;
  }
  return b.v;
}

// C[16 x Nv] tile at column n0: sum_k A(16,K) * B(K,Nv)
__device__ __forceinline__ v8f gemm16(const float* A, int lda, const float* B, int ldb,
                                      int n0, int K, int Nv, bool absB) {
  v8f acc = {};
  for (int kb = 0; kb < K; kb += 32) {
    v16h a = fragA_f32(A, lda, kb, K);
    v16h b = fragB_f32(B, ldb, n0, kb, K, Nv, absB);
    acc = wmma_f16(a, b, acc);
  }
  return acc;
}

// Scatter a D tile to row-major f32 (D layout: vgpr r -> M=r (+8 hi lanes),
// lane&15 -> N)
__device__ __forceinline__ void store_tile(float* D, int ldd, int n0, v8f acc,
                                           const float* bias) {
  int lane = threadIdx.x & 31;
  int n    = n0 + (lane & 15);
  int mb   = (lane & 16) ? 8 : 0;
  float bv = bias ? bias[n] : 0.0f;
#pragma unroll
  for (int r = 0; r < 8; ++r) D[(mb + r) * ldd + n] = acc[r] + bv;
}

// ---------------------------------------------------------------------------
// Small utility kernels
// ---------------------------------------------------------------------------
__global__ void zero_f32(float* p, int n) {
  int i = blockIdx.x * blockDim.x + threadIdx.x;
  if (i < n) p[i] = 0.0f;
}

// x (N,H,W,3) f32 -> padded (N,H+6,W+6,8) f16 with zero halo / channel pad
__global__ void pack_pad_input(const float* __restrict__ x, _Float16* __restrict__ p,
                               int N, int H, int W, int Cin) {
  int HP = H + 6, WP = W + 6;
  size_t total = (size_t)N * HP * WP * 8;
  size_t i = (size_t)blockIdx.x * blockDim.x + threadIdx.x;
  if (i >= total) return;
  int c = (int)(i & 7);
  size_t rest = i >> 3;
  int wp = (int)(rest % WP); rest /= WP;
  int hp = (int)(rest % HP);
  int n  = (int)(rest / HP);
  float v = 0.0f;
  if (hp >= 3 && hp < H + 3 && wp >= 3 && wp < W + 3 && c < Cin)
    v = x[(((size_t)n * H + (hp - 3)) * W + (wp - 3)) * Cin + c];
  p[i] = (_Float16)v;
}

__global__ void bn_finalize(const float* __restrict__ stats, const float* __restrict__ g,
                            const float* __restrict__ b, float* __restrict__ f,
                            int C, float invCnt) {
  int c = threadIdx.x;
  if (c < C) {
    float mean = stats[c] * invCnt;
    float var  = stats[C + c] * invCnt - mean * mean;
    float sc   = g[c] * rsqrtf(var + 1e-5f);
    f[c]     = sc;
    f[C + c] = b[c] - mean * sc;
  }
}

// affine(BN) + leaky_relu + 2x2 maxpool -> padded f16 buffer
// cin: (N,H,W,C) f32 ; out: (N, H/2+6, W/2+6, Cpout) zero halo / channel pad
__global__ void bn_lrelu_pool_pad(const float* __restrict__ cin,
                                  const float* __restrict__ f,
                                  _Float16* __restrict__ outp, int N, int H, int W,
                                  int C, int Cpout) {
  int Ho = H >> 1, Wo = W >> 1;
  int HP = Ho + 6, WP = Wo + 6;
  size_t total = (size_t)N * HP * WP * Cpout;
  size_t i = (size_t)blockIdx.x * blockDim.x + threadIdx.x;
  if (i >= total) return;
  int c = (int)(i % Cpout);
  size_t rest = i / Cpout;
  int wp = (int)(rest % WP); rest /= WP;
  int hp = (int)(rest % HP);
  int n  = (int)(rest / HP);
  _Float16 res = (_Float16)0.0f;
  if (hp >= 3 && hp < Ho + 3 && wp >= 3 && wp < Wo + 3 && c < C) {
    int ho = hp - 3, wo = wp - 3;
    float sc = f[c], sh = f[C + c];
    float mx = -3.4e38f;
#pragma unroll
    for (int dy = 0; dy < 2; ++dy)
#pragma unroll
      for (int dx = 0; dx < 2; ++dx) {
        size_t idx = (((size_t)n * H + (2 * ho + dy)) * W + (2 * wo + dx)) * C + c;
        float v = cin[idx] * sc + sh;
        v = (v > 0.0f) ? v : 0.01f * v;   // leaky_relu
        mx = fmaxf(mx, v);
      }
    res = (_Float16)mx;
  }
  outp[i] = res;
}

// ---------------------------------------------------------------------------
// 7x7 SAME conv, packed-tap implicit GEMM: 32/CP kw-adjacent taps per WMMA
// (K = tapInGroup*CP + channel), fully utilizing K=32.
// in: (N, H+6, W+6, CP) f16 zero-halo, w: HWIO f32, out: (N,H,W,Cout) f32.
// Unified lane addressing for all CP: base = pixel + (hiLane ? 8 : 0) halfs,
// chunk0 at compile-time off, chunk1 at off+16 halfs. Groups whose kw >= 7
// have zero B rows (their loads may read harmlessly into the next buffer).
// ---------------------------------------------------------------------------
#define CONV_WAVES 8
template <int CP, int WP, int HH>
__global__ void __launch_bounds__(256)
conv7x7_wmma(const _Float16* __restrict__ in, const float* __restrict__ w,
             const float* __restrict__ bias, float* __restrict__ out,
             float* __restrict__ stats, int N, int Cin, int Cout) {
  constexpr int W   = WP - 6;
  constexpr int HP  = HH + 6;
  constexpr int HW  = HH * W;
  constexpr int KWS = 32 / CP;          // taps packed per WMMA (4 / 2 / 1)
  constexpr int JN  = (6 + KWS) / KWS;  // kw groups per row: 2 / 4 / 7
  constexpr int NG  = 7 * JN;           // WMMA groups: 14 / 28 / 49
  __shared__ _Float16 sB[NG * 32 * 16];
  __shared__ float sRed[32];
  const int tid  = threadIdx.x;
  const int lane = tid & 31;
  const int wave = tid >> 5;
  const int ct   = blockIdx.y;

  // stage B fragments: sB[g][lane][pos]; K -> (kw = j*KWS + K/CP, c = K%CP)
  for (int e = tid; e < NG * 512; e += 256) {
    int g   = e >> 9;
    int r   = e & 511;
    int ln  = r >> 4;
    int pos = r & 15;
    int hf  = (ln >> 4) & 1;
    int n   = ct * 16 + (ln & 15);
    int k   = (pos < 8) ? (hf * 8 + pos) : (16 + hf * 8 + (pos - 8));
    int kh  = g / JN;
    int j   = g - kh * JN;
    int kw  = j * KWS + k / CP;
    int c   = k % CP;
    float v = (kw < 7 && c < Cin && n < Cout)
                  ? w[(((size_t)kh * 7 + kw) * Cin + c) * Cout + n]
                  : 0.0f;
    sB[e] = (_Float16)v;
  }
  if (tid < 32) sRed[tid] = 0.0f;
  __syncthreads();

  const int nStrips = (N * HW) >> 4;
  const int m   = lane & 15;                 // A row / D column index
  const int hi8 = (lane & 16) ? 8 : 0;       // lane-half base shift (halfs)
  const int ch  = ct * 16 + m;               // this lane's output channel
  float ssum = 0.0f, ssq = 0.0f;
  float bval = (bias && ch < Cout) ? bias[ch] : 0.0f;

  for (int strip = blockIdx.x * CONV_WAVES + wave; strip < nStrips;
       strip += gridDim.x * CONV_WAVES) {
    int p0  = strip << 4;
    int p   = p0 + m;
    int img = p / HW;                        // compile-time HW -> shifts
    int rem = p - img * HW;
    int hh  = rem / W;
    int ww  = rem - hh * W;
    const _Float16* base =
        in + (((size_t)img * HP + (hh + 3)) * WP + (ww + 3)) * CP + hi8;
    v8f acc = {};
#pragma unroll
    for (int kh = 0; kh < 7; ++kh) {
#pragma unroll
      for (int j = 0; j < JN; ++j) {
        const int off = ((kh - 3) * WP - 3) * CP + j * 32;   // compile-time
        union { v16h v; v8h h2[2]; } a;
        a.h2[0] = *reinterpret_cast<const v8h*>(base + off);
        a.h2[1] = *reinterpret_cast<const v8h*>(base + off + 16);
        v16h b =
            *reinterpret_cast<const v16h*>(&sB[((kh * JN + j) * 32 + lane) * 16]);
        acc = wmma_f16(a.v, b, acc);
      }
    }
    if (ch < Cout) {
      int mb = (lane & 16) ? 8 : 0;
#pragma unroll
      for (int r = 0; r < 8; ++r) {
        float v = acc[r] + bval;
        out[(size_t)(p0 + mb + r) * Cout + ch] = v;
        ssum += v;
        ssq  += v * v;
      }
    }
  }

  if (stats) {
    if (ch < Cout) {
      atomicAdd(&sRed[m], ssum);
      atomicAdd(&sRed[16 + m], ssq);
    }
    __syncthreads();
    if (tid < 16 && (ct * 16 + tid) < Cout) {
      atomicAdd(&stats[ct * 16 + tid], sRed[tid]);
      atomicAdd(&stats[Cout + ct * 16 + tid], sRed[16 + tid]);
    }
  }
}

// ---------------------------------------------------------------------------
// Gram: G = f f^T, f = (16 latents x 64 pixels) per image. A and B fragments
// are identical (symmetric), 2 WMMAs per image, 1 wave per image.
// xi: (512, 64, 16) f32 pixel-major, G: (512, 16, 16)
// ---------------------------------------------------------------------------
__global__ void __launch_bounds__(256) gram16(const float* __restrict__ xi,
                                              float* __restrict__ G) {
  int wave = threadIdx.x >> 5;
  int lane = threadIdx.x & 31;
  int img  = blockIdx.x * 8 + wave;
  const float* f = xi + (size_t)img * 1024;
  int idx   = lane & 15;
  int base0 = (lane & 16) ? 8 : 0;
  v8f acc = {};
#pragma unroll
  for (int kb = 0; kb < 64; kb += 32) {
    union { v16h v; _Float16 e[16]; } a;
#pragma unroll
    for (int j = 0; j < 8; ++j) {
      int k0 = kb + base0 + j;
      int k1 = kb + 16 + base0 + j;
      a.e[j]     = (_Float16)f[k0 * 16 + idx];   // f[p][l]; A(l,p)==B(p,l) source
      a.e[8 + j] = (_Float16)f[k1 * 16 + idx];
    }
    acc = wmma_f16(a.v, a.v, acc);
  }
  int mb = (lane & 16) ? 8 : 0;
  float* go = G + (size_t)img * 256;
#pragma unroll
  for (int r = 0; r < 8; ++r) go[(mb + r) * 16 + idx] = acc[r];
}

// ---------------------------------------------------------------------------
// Cholesky of 16x16 SPD, 16 threads per image (16 images per block).
// diag (512,16); off-diagonals (512,120) in np.tril_indices(16,-1) order.
// ---------------------------------------------------------------------------
__global__ void __launch_bounds__(256) chol16(const float* __restrict__ G,
                                              float* __restrict__ dg,
                                              float* __restrict__ od) {
  __shared__ float A[16][16][16];
  int il  = threadIdx.x >> 4;
  int i   = threadIdx.x & 15;
  int img = blockIdx.x * 16 + il;
  const float* g = G + (size_t)img * 256;
  for (int j = 0; j < 16; ++j) A[il][i][j] = g[i * 16 + j];
  __syncthreads();
  for (int k = 0; k < 16; ++k) {
    if (i == k) A[il][k][k] = sqrtf(fmaxf(A[il][k][k], 1e-20f));
    __syncthreads();
    if (i > k) A[il][i][k] = A[il][i][k] / A[il][k][k];
    __syncthreads();
    if (i > k)
      for (int j = k + 1; j <= i; ++j) A[il][i][j] -= A[il][i][k] * A[il][j][k];
    __syncthreads();
  }
  dg[(size_t)img * 16 + i] = A[il][i][i];
  if (i >= 1) {
    int base = i * (i - 1) / 2;
    for (int c = 0; c < i; ++c) od[(size_t)img * 120 + base + c] = A[il][i][c];
  }
}

// ---------------------------------------------------------------------------
// Sequential ODE-RGRU scan (b=16, 32 steps), one block of 8 waves.
// ODE MLP: 8 waves x 16-col WMMA tiles (16x128x128 GEMMs).
// GRU gates: 4 waves, each computes 6 gate tiles (r/z/n x input/hidden) for
// its 16 hidden columns in registers, then combines elementwise.
// ---------------------------------------------------------------------------
__global__ void __launch_bounds__(256) scan_kernel(
    const float* __restrict__ diag, const float* __restrict__ offd,
    const float* __restrict__ w0, const float* __restrict__ b0,
    const float* __restrict__ wsv, const float* __restrict__ bsv,
    const float* __restrict__ wout, const float* __restrict__ bout,
    const float* __restrict__ wx_d, const float* __restrict__ wh_d,
    const float* __restrict__ bias_d, const float* __restrict__ wx_l,
    const float* __restrict__ wh_l, const float* __restrict__ bias_l,
    const float* __restrict__ cls_w, const float* __restrict__ cls_b,
    float* __restrict__ out) {
  __shared__ float sY[16][128], sT[16][128], sH[16][128], sSum[16][128];
  __shared__ float sHd[16][64], sHl[16][64];
  __shared__ float sXd[16][16], sXl[16][120];
  const int tid  = threadIdx.x;
  const int lane = tid & 31;
  const int wave = tid >> 5;

  for (int i = tid; i < 16 * 64; i += 256) {
    (&sHd[0][0])[i] = 1.0f;
    (&sHl[0][0])[i] = 0.0f;
  }
  for (int i = tid; i < 2048; i += 256) (&sSum[0][0])[i] = 0.0f;
  __syncthreads();

  for (int t = 0; t < 32; ++t) {
    // stage x_t
    {
      int m = tid >> 4, l = tid & 15;
      sXd[m][l] = diag[((size_t)m * 32 + t) * 16 + l];
    }
    for (int i = tid; i < 1920; i += 256) {
      int m = i / 120, j = i - m * 120;
      sXl[m][j] = offd[((size_t)m * 32 + t) * 120 + j];
    }
    // y = [log(h_d), h_l]
    for (int i = tid; i < 2048; i += 256) {
      int m = i >> 7, c = i & 127;
      (&sY[0][0])[i] = (c < 64) ? logf(sHd[m][c]) : sHl[m][c - 64];
    }
    __syncthreads();

    const int n0 = wave * 16;
    // ODE MLP: h = y@w0+b0; 2x: h = tanh(h)@ws+bs; out = tanh(h)@wout+bout
    { v8f a = gemm16(&sY[0][0], 128, w0, 128, n0, 128, 128, false);
      store_tile(&sH[0][0], 128, n0, a, b0); }
    __syncthreads();
    for (int i = tid; i < 2048; i += 256) (&sT[0][0])[i] = tanhf((&sH[0][0])[i]);
    __syncthreads();
    { v8f a = gemm16(&sT[0][0], 128, wsv, 128, n0, 128, 128, false);
      store_tile(&sH[0][0], 128, n0, a, bsv); }
    __syncthreads();
    for (int i = tid; i < 2048; i += 256) (&sT[0][0])[i] = tanhf((&sH[0][0])[i]);
    __syncthreads();
    { v8f a = gemm16(&sT[0][0], 128, wsv + 128 * 128, 128, n0, 128, 128, false);
      store_tile(&sH[0][0], 128, n0, a, bsv + 128); }
    __syncthreads();
    for (int i = tid; i < 2048; i += 256) (&sT[0][0])[i] = tanhf((&sH[0][0])[i]);
    __syncthreads();
    { v8f a = gemm16(&sT[0][0], 128, wout, 128, n0, 128, 128, false);
      store_tile(&sH[0][0], 128, n0, a, bout); }
    __syncthreads();

    // y += ode(y);  h_d = exp(tanh(y[:, :64]));  h_l = y[:, 64:]
    for (int i = tid; i < 2048; i += 256) {
      int m = i >> 7, c = i & 127;
      float v = (&sY[0][0])[i] + (&sH[0][0])[i];
      (&sY[0][0])[i] = v;
      if (c < 64) sHd[m][c] = expf(tanhf(v));
      else        sHl[m][c - 64] = v;
    }
    __syncthreads();

    // ---- diagonal GRU branch (abs weights/biases, multiplicative gates) ----
    v8f ir = {}, ii = {}, in_ = {}, hr = {}, hi = {}, hn = {};
    if (wave < 4) {
      int cb = wave * 16;
      ir  = gemm16(&sXd[0][0], 16, wx_d, 192, cb,       16, 192, true);
      ii  = gemm16(&sXd[0][0], 16, wx_d, 192, 64 + cb,  16, 192, true);
      in_ = gemm16(&sXd[0][0], 16, wx_d, 192, 128 + cb, 16, 192, true);
      hr  = gemm16(&sHd[0][0], 64, wh_d, 192, cb,       64, 192, true);
      hi  = gemm16(&sHd[0][0], 64, wh_d, 192, 64 + cb,  64, 192, true);
      hn  = gemm16(&sHd[0][0], 64, wh_d, 192, 128 + cb, 64, 192, true);
    }
    __syncthreads();   // all reads of sHd done before writes
    if (wave < 4) {
      int cb = wave * 16;
      int c  = cb + (lane & 15);
      int mb = (lane & 16) ? 8 : 0;
      float br = fabsf(bias_d[c]);
      float bi = fabsf(bias_d[64 + c]);
      float bn = fabsf(bias_d[128 + c]);
#pragma unroll
      for (int r = 0; r < 8; ++r) {
        int m = mb + r;
        float rg  = sigmoidf_(br * ir[r] * hr[r]);
        float zg  = sigmoidf_(bi * ii[r] * hi[r]);
        float ng  = softplusf_(bn * in_[r] * rg * hn[r]);
        float old = sHd[m][c];
        sHd[m][c] = expf((1.0f - zg) * logf(ng) + zg * logf(old));
      }
    }
    __syncthreads();

    // ---- lower-triangular GRU branch (standard additive GRU) ----
    v8f lir = {}, lii = {}, lin = {}, lhr = {}, lhi = {}, lhn = {};
    if (wave < 4) {
      int cb = wave * 16;
      lir = gemm16(&sXl[0][0], 120, wx_l, 192, cb,       120, 192, false);
      lii = gemm16(&sXl[0][0], 120, wx_l, 192, 64 + cb,  120, 192, false);
      lin = gemm16(&sXl[0][0], 120, wx_l, 192, 128 + cb, 120, 192, false);
      lhr = gemm16(&sHl[0][0], 64, wh_l, 192, cb,        64, 192, false);
      lhi = gemm16(&sHl[0][0], 64, wh_l, 192, 64 + cb,   64, 192, false);
      lhn = gemm16(&sHl[0][0], 64, wh_l, 192, 128 + cb,  64, 192, false);
    }
    __syncthreads();
    if (wave < 4) {
      int cb = wave * 16;
      int c  = cb + (lane & 15);
      int mb = (lane & 16) ? 8 : 0;
      float br = bias_l[c], bi = bias_l[64 + c], bn = bias_l[128 + c];
#pragma unroll
      for (int r = 0; r < 8; ++r) {
        int m = mb + r;
        float lr  = sigmoidf_(lir[r] + lhr[r] + br);
        float lz  = sigmoidf_(lii[r] + lhi[r] + bi);
        float lng = tanhf(lin[r] + lr * lhn[r] + bn);
        float old = sHl[m][c];
        sHl[m][c] = lng + lz * (old - lng);
      }
    }
    __syncthreads();

    // accumulate step output [log(h_d), h_l]
    for (int i = tid; i < 2048; i += 256) {
      int m = i >> 7, c = i & 127;
      (&sSum[0][0])[i] += (c < 64) ? logf(sHd[m][c]) : sHl[m][c - 64];
    }
    __syncthreads();
  }

  // logits = mean_t(outs) @ cls_w + cls_b  (16x10)
  if (tid < 160) {
    int m = tid / 10, c = tid - m * 10;
    float acc = cls_b[c];
    for (int k = 0; k < 128; ++k)
      acc += (sSum[m][k] * (1.0f / 32.0f)) * cls_w[k * 10 + c];
    out[m * 10 + c] = acc;
  }
}

// ---------------------------------------------------------------------------
// Host launcher
// ---------------------------------------------------------------------------
extern "C" void kernel_launch(void* const* d_in, const int* in_sizes, int n_in,
                              void* d_out, int out_size, void* d_ws, size_t ws_size,
                              hipStream_t stream) {
  (void)in_sizes; (void)n_in; (void)out_size; (void)ws_size;
  const float* x      = (const float*)d_in[0];
  const float* w1     = (const float*)d_in[1];
  // conv1_b/conv2_b/conv3_b (d_in[2],[6],[10]) cancel inside BN mean-subtract
  const float* g1     = (const float*)d_in[3];
  const float* bb1    = (const float*)d_in[4];
  const float* w2     = (const float*)d_in[5];
  const float* g2     = (const float*)d_in[7];
  const float* bb2    = (const float*)d_in[8];
  const float* w3     = (const float*)d_in[9];
  const float* g3     = (const float*)d_in[11];
  const float* bb3    = (const float*)d_in[12];
  const float* w4     = (const float*)d_in[13];
  const float* b4     = (const float*)d_in[14];
  const float* ode_w0 = (const float*)d_in[15];
  const float* ode_b0 = (const float*)d_in[16];
  const float* ode_ws = (const float*)d_in[17];
  const float* ode_bs = (const float*)d_in[18];
  const float* ode_wo = (const float*)d_in[19];
  const float* ode_bo = (const float*)d_in[20];
  const float* wx_d   = (const float*)d_in[21];
  const float* wh_d   = (const float*)d_in[22];
  const float* bias_d = (const float*)d_in[23];
  const float* wx_l   = (const float*)d_in[24];
  const float* wh_l   = (const float*)d_in[25];
  const float* bias_l = (const float*)d_in[26];
  const float* cls_w  = (const float*)d_in[27];
  const float* cls_b  = (const float*)d_in[28];
  float* out = (float*)d_out;

  char* base = (char*)d_ws;
  size_t off = 0;
  auto alloc = [&](size_t bytes) -> char* {
    char* p = base + off;
    off += (bytes + 255) & ~(size_t)255;
    return p;
  };
  // Padded f16 activation buffers (3-px halo).
  // PA hosts P1 (512*70*70*8) then P3 (512*22*22*16); PB hosts P2 (512*38*38*8)
  // then P4 (512*14*14*32); CBUF hosts conv1/2/3 f32 outputs (max 67MB).
  _Float16* PA = (_Float16*)alloc((size_t)512 * 70 * 70 * 8 * 2);
  _Float16* PB = (_Float16*)alloc((size_t)512 * 38 * 38 * 8 * 2);
  float* CBUF  = (float*)alloc((size_t)512 * 64 * 64 * 8 * 4);
  float* C4    = (float*)alloc((size_t)512 * 64 * 16 * 4);
  float* G     = (float*)alloc((size_t)512 * 256 * 4);
  float* DG    = (float*)alloc((size_t)512 * 16 * 4);
  float* OD    = (float*)alloc((size_t)512 * 120 * 4);
  float* SM    = (float*)alloc(256 * 4);
  (void)alloc(256);  // tail guard for the (zero-weighted) kw==7 wrap reads
  // SM layout: s1@0(16) f1@16(16) s2@32(32) f2@64(32) s3@96(64) f3@160(64)

  zero_f32<<<1, 256, 0, stream>>>(SM, 224);

  {
    size_t tot = (size_t)512 * 70 * 70 * 8;
    pack_pad_input<<<(unsigned)((tot + 255) / 256), 256, 0, stream>>>(x, PA, 512, 64,
                                                                      64, 3);
  }
  // layer 1: conv(64x64, 3->8) + BN stats -> finalize -> pool -> padded 32x32x8
  conv7x7_wmma<8, 70, 64><<<dim3(2048, 1), 256, 0, stream>>>(PA, w1, nullptr, CBUF,
                                                             SM + 0, 512, 3, 8);
  bn_finalize<<<1, 64, 0, stream>>>(SM + 0, g1, bb1, SM + 16, 8,
                                    1.0f / (512.0f * 64.0f * 64.0f));
  {
    size_t tot = (size_t)512 * 38 * 38 * 8;
    bn_lrelu_pool_pad<<<(unsigned)((tot + 255) / 256), 256, 0, stream>>>(
        CBUF, SM + 16, PB, 512, 64, 64, 8, 8);
  }
  // layer 2: conv(32x32, 8->16) -> pool -> padded 16x16x16
  conv7x7_wmma<8, 38, 32><<<dim3(1024, 1), 256, 0, stream>>>(PB, w2, nullptr, CBUF,
                                                             SM + 32, 512, 8, 16);
  bn_finalize<<<1, 64, 0, stream>>>(SM + 32, g2, bb2, SM + 64, 16,
                                    1.0f / (512.0f * 32.0f * 32.0f));
  {
    size_t tot = (size_t)512 * 22 * 22 * 16;
    bn_lrelu_pool_pad<<<(unsigned)((tot + 255) / 256), 256, 0, stream>>>(
        CBUF, SM + 64, PA, 512, 32, 32, 16, 16);
  }
  // layer 3: conv(16x16, 16->32, two ctiles) -> pool -> padded 8x8x32
  conv7x7_wmma<16, 22, 16><<<dim3(512, 2), 256, 0, stream>>>(PA, w3, nullptr, CBUF,
                                                             SM + 96, 512, 16, 32);
  bn_finalize<<<1, 64, 0, stream>>>(SM + 96, g3, bb3, SM + 160, 32,
                                    1.0f / (512.0f * 16.0f * 16.0f));
  {
    size_t tot = (size_t)512 * 14 * 14 * 32;
    bn_lrelu_pool_pad<<<(unsigned)((tot + 255) / 256), 256, 0, stream>>>(
        CBUF, SM + 160, PB, 512, 16, 16, 32, 32);
  }
  // layer 4: conv(8x8, 32->16) + bias, no BN
  conv7x7_wmma<32, 14, 8><<<dim3(128, 1), 256, 0, stream>>>(PB, w4, b4, C4, nullptr,
                                                            512, 32, 16);
  // Gram + Cholesky
  gram16<<<64, 256, 0, stream>>>(C4, G);
  chol16<<<32, 256, 0, stream>>>(G, DG, OD);
  // sequential scan + classifier
  scan_kernel<<<1, 256, 0, stream>>>(DG, OD, ode_w0, ode_b0, ode_ws, ode_bs, ode_wo,
                                     ode_bo, wx_d, wh_d, bias_d, wx_l, wh_l, bias_l,
                                     cls_w, cls_b, out);
}
